// EdgeLearning_8839042695496
// MI455X (gfx1250) — compile-verified, matrix operand order checked by reading, not run on hardware
//
#include <hip/hip_runtime.h>

// ---------------------------------------------------------------------------
// MI455X (gfx1250) edge-MLP: gather + 3-branch GEMM + relu + GEMM
// bf16 WMMA (V_WMMA_F32_16X16X32_BF16), f32 accumulation, wave32.
// Async global<->LDS (ASYNCcnt) for node-row gathers and output stores.
// ---------------------------------------------------------------------------

typedef __attribute__((ext_vector_type(16))) __bf16 v16bf;
typedef __attribute__((ext_vector_type(8)))  __bf16 v8bf;
typedef __attribute__((ext_vector_type(4)))  __bf16 v4bf;
typedef __attribute__((ext_vector_type(8)))  float  v8f;
typedef int v4i __attribute__((vector_size(16)));   // matches builtin param V4i

#define DIM_NODE 128
#define DIM_EDGE 32
#define OUT_DIM  64
#define TILE_E   128

// LDS pitches (elements). Rows 16B-aligned; row-to-row bank stagger avoids
// 64-bank conflicts for the b128 fragment loads.
#define XPITCH   136   // bf16: 272 B rows -> 4-bank stride
#define EPITCH   48    // bf16: 96 B rows
#define MPITCH   72    // bf16: 144 B rows -> 36-bank stride
#define OPITCH   68    // f32 out staging: 272 B rows -> 4-bank stride

// Workspace layout
#define WTI_OFF 0                      // bf16 Wt_i 64x128
#define WTJ_OFF 8192                   // bf16 Wt_j 64x128
#define WTE_OFF 16384                  // bf16 Wt_e 64x32
#define WTO_OFF 18432                  // bf16 Wt_o 64x64
#define WT_TOTAL 22528                 // elements (45 KB)
#define XBF_OFF_BYTES 65536            // bf16 node table starts here

// gfx1250 async global<->LDS builtins (guarded; fallbacks compiled otherwise)
#if __has_builtin(__builtin_amdgcn_global_load_async_to_lds_b128)
#  define HAVE_ASYNC_LD 1
#else
#  define HAVE_ASYNC_LD 0
#endif
#if __has_builtin(__builtin_amdgcn_global_store_async_from_lds_b128)
#  define HAVE_ASYNC_ST 1
#else
#  define HAVE_ASYNC_ST 0
#endif

#define AS1 __attribute__((address_space(1)))
#define AS3 __attribute__((address_space(3)))

__device__ __forceinline__ void wait_async0()
{
#if HAVE_ASYNC_LD || HAVE_ASYNC_ST
#if __has_builtin(__builtin_amdgcn_s_wait_asynccnt)
    __builtin_amdgcn_s_wait_asynccnt(0);
#else
    asm volatile("s_wait_asynccnt 0x0" ::: "memory");
#endif
#endif
}

// ---------------------------------------------------------------------------
// Prep 1: convert + transpose weights (f32 [K][N]) -> bf16 [N][K] row-major.
// ---------------------------------------------------------------------------
__global__ __launch_bounds__(256) void prep_weights(
    const float* __restrict__ Wi, const float* __restrict__ Wj,
    const float* __restrict__ We, const float* __restrict__ Wo,
    __bf16* __restrict__ wt)
{
    int id = blockIdx.x * blockDim.x + threadIdx.x;
    if (id >= WT_TOTAL) return;
    float v;
    if (id < WTJ_OFF)       { int l = id;           int n = l >> 7, k = l & 127; v = Wi[k * OUT_DIM + n]; }
    else if (id < WTE_OFF)  { int l = id - WTJ_OFF; int n = l >> 7, k = l & 127; v = Wj[k * OUT_DIM + n]; }
    else if (id < WTO_OFF)  { int l = id - WTE_OFF; int n = l >> 5, k = l & 31;  v = We[k * OUT_DIM + n]; }
    else                    { int l = id - WTO_OFF; int n = l >> 6, k = l & 63;  v = Wo[k * OUT_DIM + n]; }
    wt[id] = (__bf16)v;
}

// ---------------------------------------------------------------------------
// Prep 2: node table f32 -> bf16 (halves gather L2 traffic; gathers become
// pure byte copies -> async global->LDS eligible).
// ---------------------------------------------------------------------------
__global__ __launch_bounds__(256) void prep_x(
    const float* __restrict__ xf, __bf16* __restrict__ xb, long long n4)
{
    long long i = (long long)blockIdx.x * 256 + threadIdx.x;   // float4 units
    if (i >= n4) return;
    const float4 v = ((const float4*)xf)[i];
    v4bf pk = { (__bf16)v.x, (__bf16)v.y, (__bf16)v.z, (__bf16)v.w };
    ((v4bf*)xb)[i] = pk;
}

// ---------------------------------------------------------------------------
// Fragment load matching the CDNA5 16-bit A-matrix 16x32 lane layout:
//   lane L (0-15):  row M=L,    K = {k0..k0+7} U {k0+16..k0+23}
//   lane L (16-31): row M=L-16, K = {k0+8..k0+15} U {k0+24..k0+31}
// B fragments mirror this with N in place of M (weights stored [N][K]).
// ---------------------------------------------------------------------------
__device__ __forceinline__ v16bf load_frag(const __bf16* base, int pitch, int k0, int lane)
{
    const int row  = lane & 15;
    const int half = (lane >> 4) << 3;
    const __bf16* p = base + row * pitch + k0 + half;
    union { v16bf v; v8bf h[2]; } u;
    u.h[0] = *(const v8bf*)(p);
    u.h[1] = *(const v8bf*)(p + 16);
    return u.v;
}

template<int K>
__device__ __forceinline__ void gemm_acc(const __bf16* abase, int apitch,
                                         const __bf16* wt, int lane, v8f acc[4])
{
    #pragma unroll
    for (int kc = 0; kc < K; kc += 32) {
        v16bf a = load_frag(abase, apitch, kc, lane);
        #pragma unroll
        for (int t = 0; t < 4; ++t) {
            v16bf b = load_frag(wt + t * 16 * K, K, kc, lane);
            acc[t] = __builtin_amdgcn_wmma_f32_16x16x32_bf16(
                false, a, false, b, (short)0, acc[t], false, false);
        }
    }
}

// ---------------------------------------------------------------------------
// Main kernel: one block = 128 edges, 8 waves; wave w owns rows [16w,16w+16).
// USE_BF: gather from the pre-converted bf16 node table (async if available).
// ---------------------------------------------------------------------------
template<bool USE_BF>
__global__ __launch_bounds__(256) void edge_mlp_kernel(
    const float* __restrict__ xf,
    const __bf16* __restrict__ xbf,
    const long long* __restrict__ eidx,      // [2][E] int64
    const float* __restrict__ eattr,
    const float* __restrict__ b_i, const float* __restrict__ b_j,
    const float* __restrict__ b_e, const float* __restrict__ b_o,
    const __bf16* __restrict__ wt,
    float* __restrict__ out, long long E)
{
    __shared__ alignas(16) __bf16 xbuf[TILE_E * XPITCH];   // x_i -> x_j -> mask -> out stage
    __shared__ alignas(16) __bf16 eabuf[TILE_E * EPITCH];

    const int tid  = threadIdx.x;
    const int lane = tid & 31;
    const int wave = tid >> 5;
    const long long e0 = (long long)blockIdx.x * TILE_E;

    const int col = lane & 15;                // N within 16-wide tile (C layout)
    const int rhi = (lane >> 4) << 3;         // +0 / +8 row within 16x16 tile

    // ---- gather of one x side (which = 0: src, 1: dst) ----
    auto gather_x = [&](int which) {
        if constexpr (USE_BF) {
            // 2 rows per wave per iter; per-lane 16B chunk; async global->LDS
            #pragma unroll 1
            for (int it = 0; it < TILE_E / 16; ++it) {
                const int r = it * 16 + wave * 2 + (lane >> 4);
                const long long e = e0 + r;
                const long long node = (e < E) ? eidx[(long long)which * E + e] : 0;
                const int c8 = (lane & 15) * 8;                 // bf16 elements
                const __bf16* gp = xbf + node * DIM_NODE + c8;
                __bf16* lp = &xbuf[r * XPITCH + c8];
#if HAVE_ASYNC_LD
                __builtin_amdgcn_global_load_async_to_lds_b128(
                    (AS1 v4i*)gp, (AS3 v4i*)lp, 0, 0);
#else
                *(v8bf*)lp = *(const v8bf*)gp;
#endif
            }
#if HAVE_ASYNC_LD
            wait_async0();
#endif
        } else {
            // f32 gather + convert (fallback when d_ws lacks the bf16 table)
            #pragma unroll 1
            for (int it = 0; it < TILE_E / 8; ++it) {
                const int r = it * 8 + wave;
                const long long e = e0 + r;
                const long long node = (e < E) ? eidx[(long long)which * E + e] : 0;
                const float4 v = ((const float4*)(xf + node * DIM_NODE))[lane];
                v4bf pk = { (__bf16)v.x, (__bf16)v.y, (__bf16)v.z, (__bf16)v.w };
                *(v4bf*)(&xbuf[r * XPITCH + lane * 4]) = pk;
            }
        }
    };

    // ---- phase 0: gather x[src] + edge_attr ----
    gather_x(0);
    #pragma unroll 1
    for (int f = tid; f < TILE_E * (DIM_EDGE / 4); f += 256) {
        const int r  = f >> 3;
        const int c4 = f & 7;
        const long long e = e0 + r;
        float4 v = {0.f, 0.f, 0.f, 0.f};
        if (e < E) v = ((const float4*)(eattr + e * DIM_EDGE))[c4];
        v4bf pk = { (__bf16)v.x, (__bf16)v.y, (__bf16)v.z, (__bf16)v.w };
        *(v4bf*)(&eabuf[r * EPITCH + c4 * 4]) = pk;
    }
    __syncthreads();

    const __bf16* astrip = xbuf + (wave * 16) * XPITCH;

    float bi[4], bj[4], be[4], bo[4];
    #pragma unroll
    for (int t = 0; t < 4; ++t) {
        bi[t] = b_i[t * 16 + col]; bj[t] = b_j[t * 16 + col];
        be[t] = b_e[t * 16 + col]; bo[t] = b_o[t * 16 + col];
    }

    const v8f vzero = {0.f, 0.f, 0.f, 0.f, 0.f, 0.f, 0.f, 0.f};
    v8f hsum[4];

    // ---- GEMM1: hsum = relu(x_i @ W_i + b_i) ----
    {
        v8f acc[4] = {vzero, vzero, vzero, vzero};
        gemm_acc<DIM_NODE>(astrip, XPITCH, wt + WTI_OFF, lane, acc);
        #pragma unroll
        for (int t = 0; t < 4; ++t)
            #pragma unroll
            for (int g = 0; g < 8; ++g)
                hsum[t][g] = fmaxf(acc[t][g] + bi[t], 0.f);
    }
    __syncthreads();                       // all waves done reading x_i

    // ---- phase 1: gather x[dst] into the same buffer ----
    gather_x(1);
    __syncthreads();

    // ---- GEMM2: hsum += relu(x_j @ W_j + b_j) ----
    {
        v8f acc[4] = {vzero, vzero, vzero, vzero};
        gemm_acc<DIM_NODE>(astrip, XPITCH, wt + WTJ_OFF, lane, acc);
        #pragma unroll
        for (int t = 0; t < 4; ++t)
            #pragma unroll
            for (int g = 0; g < 8; ++g)
                hsum[t][g] += fmaxf(acc[t][g] + bj[t], 0.f);
    }
    // ---- GEMM3: hsum += relu(edge_attr @ W_e + b_e) ----
    {
        v8f acc[4] = {vzero, vzero, vzero, vzero};
        gemm_acc<DIM_EDGE>(eabuf + (wave * 16) * EPITCH, EPITCH, wt + WTE_OFF, lane, acc);
        #pragma unroll
        for (int t = 0; t < 4; ++t)
            #pragma unroll
            for (int g = 0; g < 8; ++g)
                hsum[t][g] += fmaxf(acc[t][g] + be[t], 0.f);
    }
    __syncthreads();                       // all waves done reading xbuf (x_j)

    // ---- mask = relu(hsum) -> bf16, overlay into xbuf with MPITCH layout ----
    __bf16* mbuf = xbuf;
    #pragma unroll
    for (int t = 0; t < 4; ++t)
        #pragma unroll
        for (int g = 0; g < 8; ++g) {
            const int row = wave * 16 + g + rhi;
            mbuf[row * MPITCH + t * 16 + col] = (__bf16)fmaxf(hsum[t][g], 0.f);
        }
    // same-wave RAW on own strip only; compiler inserts s_wait_dscnt.

    // ---- GEMM4: out = relu(mask @ W_o + b_o) ----
    v8f acc[4] = {vzero, vzero, vzero, vzero};
    gemm_acc<OUT_DIM>(mbuf + (wave * 16) * MPITCH, MPITCH, wt + WTO_OFF, lane, acc);

#if HAVE_ASYNC_ST
    // Stage result tile (f32) in LDS, then stream it out with contiguous
    // 256B-per-row async stores (ASYNCcnt; drained by S_ENDPGM's wait-idle).
    __syncthreads();                       // mask strips overlap other waves
    float* obuf = (float*)xbuf;            // 128 x OPITCH overlay (34816 B)
    #pragma unroll
    for (int t = 0; t < 4; ++t)
        #pragma unroll
        for (int g = 0; g < 8; ++g) {
            const int row = wave * 16 + g + rhi;
            obuf[row * OPITCH + t * 16 + col] = fmaxf(acc[t][g] + bo[t], 0.f);
        }
    __syncthreads();
    #pragma unroll 1
    for (int q = tid; q < TILE_E * (OUT_DIM / 4); q += 256) {  // 16B chunks
        const int row = q >> 4;
        const int c4  = q & 15;
        const long long e = e0 + row;
        if (e < E) {
            const float* lp = obuf + row * OPITCH + c4 * 4;
            float* gp = out + e * OUT_DIM + c4 * 4;
            __builtin_amdgcn_global_store_async_from_lds_b128(
                (AS1 v4i*)gp, (AS3 v4i*)lp, 0, 0);
        }
    }
#else
    #pragma unroll
    for (int t = 0; t < 4; ++t)
        #pragma unroll
        for (int g = 0; g < 8; ++g) {
            const long long e = e0 + wave * 16 + g + rhi;
            if (e < E)
                out[e * OUT_DIM + t * 16 + col] = fmaxf(acc[t][g] + bo[t], 0.f);
        }
#endif
}

// ---------------------------------------------------------------------------
// Launch
// ---------------------------------------------------------------------------
extern "C" void kernel_launch(void* const* d_in, const int* in_sizes, int n_in,
                              void* d_out, int out_size, void* d_ws, size_t ws_size,
                              hipStream_t stream)
{
    const float*     x     = (const float*)d_in[0];
    const long long* eidx  = (const long long*)d_in[1];   // int64 [2][E]
    const float*     eattr = (const float*)d_in[2];
    const float*     W_i   = (const float*)d_in[3];
    const float*     b_i   = (const float*)d_in[4];
    const float*     W_j   = (const float*)d_in[5];
    const float*     b_j   = (const float*)d_in[6];
    const float*     W_e   = (const float*)d_in[7];
    const float*     b_e   = (const float*)d_in[8];
    const float*     W_o   = (const float*)d_in[9];
    const float*     b_o   = (const float*)d_in[10];

    const long long E = in_sizes[2] / DIM_EDGE;           // edge_attr elems / 32
    __bf16* wt  = (__bf16*)d_ws;
    __bf16* xbf = (__bf16*)((char*)d_ws + XBF_OFF_BYTES);

    const size_t need = (size_t)XBF_OFF_BYTES + (size_t)in_sizes[0] * sizeof(__bf16);
    const bool use_bf = (ws_size >= need);                // deterministic branch

    prep_weights<<<(WT_TOTAL + 255) / 256, 256, 0, stream>>>(W_i, W_j, W_e, W_o, wt);

    const int nblocks = (int)((E + TILE_E - 1) / TILE_E);
    if (use_bf) {
        const long long n4 = (long long)in_sizes[0] / 4;  // float4 units
        prep_x<<<(int)((n4 + 255) / 256), 256, 0, stream>>>(x, xbf, n4);
        edge_mlp_kernel<true><<<nblocks, 256, 0, stream>>>(
            x, xbf, eidx, eattr, b_i, b_j, b_e, b_o, wt, (float*)d_out, E);
    } else {
        edge_mlp_kernel<false><<<nblocks, 256, 0, stream>>>(
            x, xbf, eidx, eattr, b_i, b_j, b_e, b_o, wt, (float*)d_out, E);
    }
}
